// TrSFRNNAgent_40226663694846
// MI455X (gfx1250) — compile-verified, matrix operand order checked by reading, not run on hardware
//
#include <hip/hip_runtime.h>
#include <hip/hip_bf16.h>
#include <math.h>

typedef __attribute__((ext_vector_type(16))) _Float16 v16h;
typedef __attribute__((ext_vector_type(8)))  _Float16 v8h;
typedef __attribute__((ext_vector_type(8)))  float    v8f;

#define TPB 256

// ---------------------------------------------------------------------------
// Kernel 0: repack fp32 weights [K,N] -> f16 WMMA B-fragment layout:
//   dst[((nt*KT + kt)*32 + lane)*16 + j] = W[k, nt*16 + (lane&15)]
//   k = kt*32 + (j>>3)*16 + ((lane&16)?8:0) + (j&7)   (zero for k >= K)
// Main kernel then loads a B fragment as one aligned 32B lane-contiguous load.
// ---------------------------------------------------------------------------
__global__ __launch_bounds__(TPB)
void pack_w(const float* __restrict__ src, _Float16* __restrict__ dst,
            int K, int N, int KT)
{
    int i = blockIdx.x * TPB + threadIdx.x;
    int total = (N >> 4) * KT * 512;
    if (i >= total) return;
    int j    = i & 15;
    int lane = (i >> 4) & 31;
    int r    = i >> 9;            // r = nt*KT + kt
    int kt   = r % KT;
    int nt   = r / KT;
    int k = kt * 32 + ((j >> 3) << 4) + ((lane & 16) ? 8 : 0) + (j & 7);
    int n = (nt << 4) + (lane & 15);
    dst[i] = (k < K) ? (_Float16)src[(size_t)k * N + n] : (_Float16)0.f;
}

// ---------------------------------------------------------------------------
// Fragment helpers (wave32, V_WMMA_F32_16X16X32_F16 layouts per CDNA5 ISA 7.12.2)
// ---------------------------------------------------------------------------
__device__ inline v16h load_afrag(const _Float16* p0, int strideK, int kt, int lane)
{
    // A 16x32 f16: lane&15 = row M; lanes 0-15 K base +0, lanes 16-31 K base +8.
    const _Float16* p = p0 + (lane & 15) * strideK + kt * 32 + ((lane & 16) ? 8 : 0);
    v8h lo = *(const v8h*)p;          // K = base+0..7   (ds_load_b128)
    v8h hi = *(const v8h*)(p + 16);   // K = base+16..23
    v16h a;
#pragma unroll
    for (int j = 0; j < 8; ++j) { a[j] = lo[j]; a[j + 8] = hi[j]; }
    return a;
}

template <int KT>
__device__ inline v8f gemm_frag(const _Float16* aLds, int strideK,
                                const _Float16* __restrict__ wp, int nt, int lane)
{
    v8f acc = {};
#pragma unroll
    for (int kt = 0; kt < KT; ++kt) {
        v16h a = load_afrag(aLds, strideK, kt, lane);
        v16h b = *(const v16h*)(wp + (((size_t)nt * KT + kt) * 32 + lane) * 16);
        acc = __builtin_amdgcn_wmma_f32_16x16x32_f16(false, a, false, b,
                                                     (short)0, acc, false, false);
    }
    return acc;
}

__device__ inline void store_frag_f32(float* dst, int strideN, v8f acc,
                                      const float* __restrict__ bias,
                                      int nt, int lane, bool leaky)
{
    int col = nt * 16 + (lane & 15);
    float bv = bias[col];
    int rb = (lane & 16) ? 8 : 0;
#pragma unroll
    for (int j = 0; j < 8; ++j) {
        float v = acc[j] + bv;
        if (leaky) v = v > 0.f ? v : 0.01f * v;
        dst[(rb + j) * strideN + col] = v;
    }
}

__device__ inline void store_frag_f16(_Float16* dst, int strideN, v8f acc,
                                      const float* __restrict__ bias,
                                      int nt, int lane, bool leaky)
{
    int col = nt * 16 + (lane & 15);
    float bv = bias[col];
    int rb = (lane & 16) ? 8 : 0;
#pragma unroll
    for (int j = 0; j < 8; ++j) {
        float v = acc[j] + bv;
        if (leaky) v = v > 0.f ? v : 0.01f * v;
        dst[(rb + j) * strideN + col] = (_Float16)v;
    }
}

// ---------------------------------------------------------------------------
// LDS layout (byte offsets); total 159104 B. Kept < 160KB so TWO blocks can be
// resident per WGP (2 x 155.4KB <= 320KB CDNA5 LDS) to overlap barrier stalls.
// ---------------------------------------------------------------------------
#define S_ALLY   0u        // f16 [7][16][96]   21504
#define S_ENEMY  21504u    // f16 [10][16][96]  30720
#define S_OWN    52224u    // f16 [16][96]       3072
#define S_QH     55296u    // f16 [16][512]     16384
#define S_SCR    71680u    // f32 [16][512]     32768  (entity scratch, buf 0)
#define S_ENG    104448u   // f32 [16][4][17]    4352  (energies -> scores)
#define S_ATA    108800u   // f32 [16][256]     16384  (ally attn accum)
#define S_ATE    125184u   // f32 [16][256]     16384  (enemy attn accum)
#define S_OWNF   141568u   // f32 [16][256]     16384  (own feature)
#define S_RED    157952u   // f32 [16][8][2]     1024  (LN reduction)
#define S_STAT   158976u   // f32 [16][2]         128  (LN mean/rstd)
#define SMEM_BYTES 159104u
// Overlays (lifetimes disjoint):
//   key-GEMM double buffer 1 -> S_ATA..S_ATE (32KB, zeroed after key loop)
//   X1 f16 [16][768] -> S_SCR ;  X2 f16 [16][512] -> S_QH
//   X3 f16 [16][512] -> S_ALLY;  Y  f32 [16][256] -> S_ATA

__device__ inline float concatv(const float* ownf, const float* ata,
                                const float* ate, int row, int c)
{
    if (c < 256) return ownf[row * 256 + c];
    if (c < 512) return ata[row * 256 + (c - 256)];
    return ate[row * 256 + (c - 512)];
}

// 256-thread energy: 4 threads per (row,head), 32-elem vectorized partial dot,
// combined with ds_add_f32 into the zero-initialized energy buffer.
__device__ inline void energy_accum(const _Float16* qh, const float* kbuf,
                                    float* eng, int t, int slot)
{
    const float escale = 0.088388347648318447f;  // 1/sqrt(128)
    int sub = t & 3, p = t >> 2;                 // p in [0,64)
    int row = p >> 2, h = p & 3;
    const _Float16* qp = qh + row * 512 + h * 128 + sub * 32;
    const float*    kp = kbuf + row * 512 + h * 128 + sub * 32;
    float s = 0.f;
#pragma unroll
    for (int c = 0; c < 4; ++c) {
        v8h qv = *(const v8h*)(qp + c * 8);      // ds_load_b128
#pragma unroll
        for (int j = 0; j < 8; ++j) s += (float)qv[j] * kp[c * 8 + j];
    }
    atomicAdd(&eng[(row * 4 + h) * 17 + slot], s * escale);
}

__device__ inline void attn_accum(float* dstAttn, const float* vbuf,
                                  const float* eng, int t, int slot)
{
    for (int idx = t; idx < 4096; idx += TPB) {
        int row = idx >> 8, c = idx & 255, h = c >> 6;
        dstAttn[idx] += eng[(row * 4 + h) * 17 + slot] * vbuf[idx];
    }
}

// ---------------------------------------------------------------------------
// Kernel 1: fully fused agent forward. One block = 16 batch rows, 8 waves.
// ---------------------------------------------------------------------------
__global__ __launch_bounds__(TPB)
void agent_fused(const float* __restrict__ own_obs,
                 const float* __restrict__ ally,
                 const float* __restrict__ enemy,
                 const _Float16* __restrict__ WqP,  const float* __restrict__ bq,
                 const _Float16* __restrict__ WakP, const float* __restrict__ bak,
                 const _Float16* __restrict__ WavP, const float* __restrict__ bav,
                 const _Float16* __restrict__ WekP, const float* __restrict__ bek,
                 const _Float16* __restrict__ WevP, const float* __restrict__ bev,
                 const _Float16* __restrict__ WownP,const float* __restrict__ bown,
                 const float* __restrict__ g1, const float* __restrict__ beta1,
                 const float* __restrict__ g2, const float* __restrict__ beta2,
                 const _Float16* __restrict__ Wf1P, const float* __restrict__ bf1,
                 const _Float16* __restrict__ Wf2P, const float* __restrict__ bf2,
                 const _Float16* __restrict__ Wf3P, const float* __restrict__ bf3,
                 float* __restrict__ out)
{
    extern __shared__ char smem[];
    _Float16* allyh  = (_Float16*)(smem + S_ALLY);
    _Float16* enemyh = (_Float16*)(smem + S_ENEMY);
    _Float16* ownh   = (_Float16*)(smem + S_OWN);
    _Float16* qh     = (_Float16*)(smem + S_QH);
    float*    scr    = (float*)(smem + S_SCR);
    float*    eng    = (float*)(smem + S_ENG);
    float*    ata    = (float*)(smem + S_ATA);
    float*    ate    = (float*)(smem + S_ATE);
    float*    ownf   = (float*)(smem + S_OWNF);
    float*    red    = (float*)(smem + S_RED);
    float*    stat   = (float*)(smem + S_STAT);
    _Float16* x1 = (_Float16*)(smem + S_SCR);   // overlay
    _Float16* x2 = (_Float16*)(smem + S_QH);    // overlay
    _Float16* x3 = (_Float16*)(smem + S_ALLY);  // overlay
    float*    yb = (float*)(smem + S_ATA);      // overlay

    const int t    = threadIdx.x;
    const int lane = t & 31;
    const int w    = t >> 5;
    const int base = blockIdx.x * 16;

    // ---- P0: stage inputs to LDS as f16 (K padded to 96), zero energy buffer
    for (int idx = t; idx < 16 * 96; idx += TPB) {
        int r = idx / 96, k = idx % 96;
        ownh[idx] = (_Float16)own_obs[(size_t)(base + r) * 96 + k];
    }
    for (int idx = t; idx < 7 * 16 * 96; idx += TPB) {
        int k = idx % 96, re = idx / 96, r = re & 15, e = re >> 4;
        float v = (k < 80) ? ally[((size_t)(base + r) * 7 + e) * 80 + k] : 0.f;
        allyh[idx] = (_Float16)v;
    }
    for (int idx = t; idx < 10 * 16 * 96; idx += TPB) {
        int k = idx % 96, re = idx / 96, r = re & 15, e = re / 16;
        float v = (k < 86) ? enemy[((size_t)(base + r) * 10 + e) * 86 + k] : 0.f;
        enemyh[idx] = (_Float16)v;
    }
    for (int idx = t; idx < 16 * 4 * 17; idx += TPB) eng[idx] = 0.f;
    __syncthreads();

    // ---- P1: query [16,512] (f16) + own_feature [16,256] (f32); 48 frags / 8 waves
#pragma unroll
    for (int i = 0; i < 6; ++i) {
        int q = w * 6 + i;
        if (q < 32) {
            v8f acc = gemm_frag<3>(ownh, 96, WqP, q, lane);
            store_frag_f16(qh, 512, acc, bq, q, lane, false);
        } else {
            int nt = q - 32;
            v8f acc = gemm_frag<3>(ownh, 96, WownP, nt, lane);
            store_frag_f32(ownf, 256, acc, bown, nt, lane, false);
        }
    }
    __syncthreads();

    // ---- P2: 17 entity key GEMMs, double-buffered (buf1 borrows attn region);
    //          energy(g-1) overlaps GEMM(g); one barrier per entity.
    {
        float* kbuf0 = scr;
        float* kbuf1 = (float*)(smem + S_ATA);   // 32KB contiguous (ata+ate)
        for (int g = 0; g < 17; ++g) {
            const _Float16* aSrc = (g < 7) ? (allyh + g * 16 * 96)
                                           : (enemyh + (g - 7) * 16 * 96);
            const _Float16* wK = (g < 7) ? WakP : WekP;
            const float*    bK = (g < 7) ? bak  : bek;
            float* kb = (g & 1) ? kbuf1 : kbuf0;
#pragma unroll
            for (int i = 0; i < 4; ++i) {
                int nt = w * 4 + i;
                v8f acc = gemm_frag<3>(aSrc, 96, wK, nt, lane);
                store_frag_f32(kb, 512, acc, bK, nt, lane, false);
            }
            if (g > 0) energy_accum(qh, ((g - 1) & 1) ? kbuf1 : kbuf0, eng, t, g - 1);
            __syncthreads();
        }
        energy_accum(qh, kbuf0, eng, t, 16);
        // attn-accumulator region (== kbuf1) is free now: zero it
        for (int idx = t; idx < 4096; idx += TPB) { ata[idx] = 0.f; ate[idx] = 0.f; }
        __syncthreads();
    }

    // ---- P3: softmax over allies [0,7) and enemies [7,17) separately
    if (t < 64) {
        int row = t >> 2, h = t & 3;
        float* ev = eng + (row * 4 + h) * 17;
        float m = ev[0];
        for (int i = 1; i < 7; ++i) m = fmaxf(m, ev[i]);
        float s = 0.f;
        for (int i = 0; i < 7; ++i) { float x = __expf(ev[i] - m); ev[i] = x; s += x; }
        float inv = 1.f / s;
        for (int i = 0; i < 7; ++i) ev[i] *= inv;
        m = ev[7];
        for (int i = 8; i < 17; ++i) m = fmaxf(m, ev[i]);
        s = 0.f;
        for (int i = 7; i < 17; ++i) { float x = __expf(ev[i] - m); ev[i] = x; s += x; }
        inv = 1.f / s;
        for (int i = 7; i < 17; ++i) ev[i] *= inv;
    }
    __syncthreads();

    // ---- P4: 17 entity value GEMMs, double-buffered inside scr (2x16KB);
    //          weighted accumulation(g-1) overlaps GEMM(g).
    {
        float* vbuf0 = scr;
        float* vbuf1 = scr + 4096;
        for (int g = 0; g < 17; ++g) {
            const _Float16* aSrc = (g < 7) ? (allyh + g * 16 * 96)
                                           : (enemyh + (g - 7) * 16 * 96);
            const _Float16* wV = (g < 7) ? WavP : WevP;
            const float*    bV = (g < 7) ? bav  : bev;
            float* vb = (g & 1) ? vbuf1 : vbuf0;
#pragma unroll
            for (int i = 0; i < 2; ++i) {
                int nt = w * 2 + i;
                v8f acc = gemm_frag<3>(aSrc, 96, wV, nt, lane);
                store_frag_f32(vb, 256, acc, bV, nt, lane, false);
            }
            if (g > 0)
                attn_accum((g - 1) < 7 ? ata : ate,
                           ((g - 1) & 1) ? vbuf1 : vbuf0, eng, t, g - 1);
            __syncthreads();
        }
        attn_accum(ate, vbuf0, eng, t, 16);
        __syncthreads();
    }

    // ---- P5: concat(own|ally|enemy) + LayerNorm1 -> X1 f16 [16][768]
    if (t < 128) {
        int row = t >> 3, sub = t & 7;
        float s1 = 0.f, s2 = 0.f;
        for (int q = 0; q < 96; ++q) {
            float v = concatv(ownf, ata, ate, row, sub * 96 + q);
            s1 += v; s2 += v * v;
        }
        red[(row * 8 + sub) * 2 + 0] = s1;
        red[(row * 8 + sub) * 2 + 1] = s2;
    }
    __syncthreads();
    if (t < 16) {
        float s1 = 0.f, s2 = 0.f;
        for (int sub = 0; sub < 8; ++sub) {
            s1 += red[(t * 8 + sub) * 2 + 0];
            s2 += red[(t * 8 + sub) * 2 + 1];
        }
        float m = s1 * (1.f / 768.f);
        float var = s2 * (1.f / 768.f) - m * m;
        stat[t * 2 + 0] = m;
        stat[t * 2 + 1] = rsqrtf(var + 1e-5f);
    }
    __syncthreads();
    if (t < 128) {
        int row = t >> 3, sub = t & 7;
        float m = stat[row * 2], rs = stat[row * 2 + 1];
        for (int q = 0; q < 96; ++q) {
            int c = sub * 96 + q;
            float v = concatv(ownf, ata, ate, row, c);
            x1[row * 768 + c] = (_Float16)((v - m) * rs * g1[c] + beta1[c]);
        }
    }
    __syncthreads();

    // ---- P6: FC1 768->512 + LeakyReLU -> X2 f16
#pragma unroll
    for (int i = 0; i < 4; ++i) {
        int nt = w * 4 + i;
        v8f acc = gemm_frag<24>(x1, 768, Wf1P, nt, lane);
        store_frag_f16(x2, 512, acc, bf1, nt, lane, true);
    }
    __syncthreads();

    // ---- P7: FC2 512->512 + LeakyReLU -> X3 f16
#pragma unroll
    for (int i = 0; i < 4; ++i) {
        int nt = w * 4 + i;
        v8f acc = gemm_frag<16>(x2, 512, Wf2P, nt, lane);
        store_frag_f16(x3, 512, acc, bf2, nt, lane, true);
    }
    __syncthreads();

    // ---- P8: FC3 512->256 + LeakyReLU -> Y f32
#pragma unroll
    for (int i = 0; i < 2; ++i) {
        int nt = w * 2 + i;
        v8f acc = gemm_frag<16>(x3, 512, Wf3P, nt, lane);
        store_frag_f32(yb, 256, acc, bf3, nt, lane, true);
    }
    __syncthreads();

    // ---- P9: LayerNorm2 + tanh * 0.1 -> global out
    if (t < 128) {
        int row = t >> 3, sub = t & 7;
        float s1 = 0.f, s2 = 0.f;
        for (int q = 0; q < 32; ++q) {
            float v = yb[row * 256 + sub * 32 + q];
            s1 += v; s2 += v * v;
        }
        red[(row * 8 + sub) * 2 + 0] = s1;
        red[(row * 8 + sub) * 2 + 1] = s2;
    }
    __syncthreads();
    if (t < 16) {
        float s1 = 0.f, s2 = 0.f;
        for (int sub = 0; sub < 8; ++sub) {
            s1 += red[(t * 8 + sub) * 2 + 0];
            s2 += red[(t * 8 + sub) * 2 + 1];
        }
        float m = s1 * (1.f / 256.f);
        float var = s2 * (1.f / 256.f) - m * m;
        stat[t * 2 + 0] = m;
        stat[t * 2 + 1] = rsqrtf(var + 1e-5f);
    }
    __syncthreads();
    for (int idx = t; idx < 4096; idx += TPB) {
        int row = idx >> 8, c = idx & 255;
        float v = (yb[idx] - stat[row * 2]) * stat[row * 2 + 1] * g2[c] + beta2[c];
        out[(size_t)(base + row) * 256 + c] = tanhf(v) * 0.1f;
    }
}

// ---------------------------------------------------------------------------
extern "C" void kernel_launch(void* const* d_in, const int* in_sizes, int n_in,
                              void* d_out, int out_size, void* d_ws, size_t ws_size,
                              hipStream_t stream)
{
    (void)in_sizes; (void)n_in; (void)out_size; (void)ws_size;
    const float* own   = (const float*)d_in[0];
    const float* ally  = (const float*)d_in[1];
    const float* enemy = (const float*)d_in[2];
    const float* Wq    = (const float*)d_in[3];
    const float* bq    = (const float*)d_in[4];
    const float* Wak   = (const float*)d_in[5];
    const float* bak   = (const float*)d_in[6];
    const float* Wav   = (const float*)d_in[7];
    const float* bav   = (const float*)d_in[8];
    const float* Wek   = (const float*)d_in[9];
    const float* bek   = (const float*)d_in[10];
    const float* Wev   = (const float*)d_in[11];
    const float* bev   = (const float*)d_in[12];
    const float* Wown  = (const float*)d_in[13];
    const float* bown  = (const float*)d_in[14];
    const float* g1    = (const float*)d_in[15];
    const float* beta1 = (const float*)d_in[16];
    const float* g2    = (const float*)d_in[17];
    const float* beta2 = (const float*)d_in[18];
    const float* Wf1   = (const float*)d_in[19];
    const float* bf1   = (const float*)d_in[20];
    const float* Wf2   = (const float*)d_in[21];
    const float* bf2   = (const float*)d_in[22];
    const float* Wf3   = (const float*)d_in[23];
    const float* bf3   = (const float*)d_in[24];
    float* out = (float*)d_out;

    _Float16* ws = (_Float16*)d_ws;
    // packed weight offsets in halves; size = (N/16)*KT*512
    _Float16* WqP   = ws + 0;       // 96x512  KT=3 : 49152
    _Float16* WakP  = ws + 49152;   // 80x512  KT=3 : 49152
    _Float16* WavP  = ws + 98304;   // 80x256  KT=3 : 24576
    _Float16* WekP  = ws + 122880;  // 86x512  KT=3 : 49152
    _Float16* WevP  = ws + 172032;  // 86x256  KT=3 : 24576
    _Float16* WownP = ws + 196608;  // 96x256  KT=3 : 24576
    _Float16* Wf1P  = ws + 221184;  // 768x512 KT=24: 393216
    _Float16* Wf2P  = ws + 614400;  // 512x512 KT=16: 262144
    _Float16* Wf3P  = ws + 876544;  // 512x256 KT=16: 131072

    struct PackJob { const float* s; _Float16* d; int K, N, KT; };
    const PackJob jobs[9] = {
        { Wq,   WqP,   96, 512, 3 }, { Wak,  WakP,  80, 512, 3 },
        { Wav,  WavP,  80, 256, 3 }, { Wek,  WekP,  86, 512, 3 },
        { Wev,  WevP,  86, 256, 3 }, { Wown, WownP, 96, 256, 3 },
        { Wf1,  Wf1P, 768, 512, 24 },{ Wf2,  Wf2P, 512, 512, 16 },
        { Wf3,  Wf3P, 512, 256, 16 },
    };
    for (int i = 0; i < 9; ++i) {
        int total = (jobs[i].N / 16) * jobs[i].KT * 512;
        pack_w<<<(total + TPB - 1) / TPB, TPB, 0, stream>>>(
            jobs[i].s, jobs[i].d, jobs[i].K, jobs[i].N, jobs[i].KT);
    }

    agent_fused<<<16384 / 16, TPB, SMEM_BYTES, stream>>>(
        own, ally, enemy,
        WqP, bq, WakP, bak, WavP, bav, WekP, bek, WevP, bev, WownP, bown,
        g1, beta1, g2, beta2, Wf1P, bf1, Wf2P, bf2, Wf3P, bf3, out);
}
